// DotProductAttention_23931557773466
// MI455X (gfx1250) — matrix operand order repaired and checked by reading, not compiled, not run on hardware
//
#include <hip/hip_runtime.h>

typedef __attribute__((ext_vector_type(16))) _Float16 v16h;
typedef __attribute__((ext_vector_type(8)))  float    v8f;

constexpr int DH  = 128;  // head dim
constexpr int BQ  = 64;   // q rows per workgroup
constexpr int BK  = 32;   // keys per tile
constexpr int NW  = 4;    // waves per workgroup
constexpr int NT  = 128;  // threads per workgroup

__launch_bounds__(NT)
__global__ void fattn_f16wmma_kernel(const float* __restrict__ Q,
                                     const float* __restrict__ K,
                                     const float* __restrict__ V,
                                     const long long* __restrict__ vlen,
                                     float* __restrict__ Out,
                                     int Lq, int Lk)
{
    __shared__ _Float16 sK [BK][DH];      // K tile, f16, [key][dim]
    __shared__ _Float16 sVt[DH][BK];      // V tile, f16, transposed: [dim][key]
    __shared__ _Float16 sP [NW][16][BK];  // per-wave P scratch (C->A relayout)

    const int tid  = threadIdx.x;
    const int wave = tid >> 5;
    const int lane = tid & 31;
    const int hf   = lane >> 4;   // 0: lanes 0-15, 1: lanes 16-31
    const int ln   = lane & 15;

    const int nqt = Lq / BQ;
    const int b   = blockIdx.x / nqt;
    const int q0  = (blockIdx.x % nqt) * BQ;

    const long long vl = vlen[b];
    const float scale = 0.088388347648318447f;  // 1/sqrt(128)

    // ---- Q fragments (A layout, 16x32 f16, 4 chunks along d) ----
    const int qrow = q0 + wave * 16 + ln;
    const float* qptr = Q + ((long long)b * Lq + qrow) * DH;
    v16h qf[4];
#pragma unroll
    for (int c = 0; c < 4; ++c) {
#pragma unroll
        for (int h = 0; h < 16; ++h) {
            const int v = h >> 1, p = h & 1;
            const int k = 32*c + (v & 3)*2 + p + hf*8 + (v >> 2)*16;
            qf[c][h] = (_Float16)qptr[k];
        }
    }

    // ---- accumulators / softmax state ----
    float m_i[8], l_i[8];
    v8f o[8];
#pragma unroll
    for (int r = 0; r < 8; ++r) { m_i[r] = -3.0e38f; l_i[r] = 0.0f; }
#pragma unroll
    for (int t = 0; t < 8; ++t)
#pragma unroll
        for (int e = 0; e < 8; ++e) o[t][e] = 0.0f;

    const float4* K4 = (const float4*)K;
    const float4* V4 = (const float4*)V;
    const int nkt = Lk / BK;

    for (int kt = 0; kt < nkt; ++kt) {
        // ---- cooperative tile load: K -> sK (f16), V -> sVt (f16, transposed) ----
        const long long base = ((long long)b * Lk + (long long)kt * BK) * DH / 4;
        for (int i = tid; i < BK * DH / 4; i += NT) {
            const int key = (i << 2) / DH;
            const int dd  = (i << 2) & (DH - 1);
            const float4 kv = K4[base + i];
            sK[key][dd+0] = (_Float16)kv.x;
            sK[key][dd+1] = (_Float16)kv.y;
            sK[key][dd+2] = (_Float16)kv.z;
            sK[key][dd+3] = (_Float16)kv.w;
            const float4 vv = V4[base + i];
            sVt[dd+0][key] = (_Float16)vv.x;
            sVt[dd+1][key] = (_Float16)vv.y;
            sVt[dd+2][key] = (_Float16)vv.z;
            sVt[dd+3][key] = (_Float16)vv.w;
        }
        if (kt + 1 < nkt) {  // global_prefetch_b8 of next K/V tile
            const long long nb = ((long long)b * Lk + (long long)(kt + 1) * BK) * DH;
            __builtin_prefetch(&K[nb], 0, 1);
            __builtin_prefetch(&V[nb], 0, 1);
        }
        __syncthreads();

        // ---- S = Q * K^T (16x32 per wave = two 16x16 C tiles) ----
        v8f c0, c1;
#pragma unroll
        for (int e = 0; e < 8; ++e) { c0[e] = 0.0f; c1[e] = 0.0f; }
#pragma unroll
        for (int c = 0; c < 4; ++c) {
            v16h b0, b1;
#pragma unroll
            for (int h = 0; h < 16; ++h) {
                const int kd = 32*c + (h & 7) + 8*hf + 16*(h >> 3);
                b0[h] = sK[ln][kd];
                b1[h] = sK[16 + ln][kd];
            }
            c0 = __builtin_amdgcn_wmma_f32_16x16x32_f16(false, qf[c], false, b0,
                                                        (short)0, c0, false, false);
            c1 = __builtin_amdgcn_wmma_f32_16x16x32_f16(false, qf[c], false, b1,
                                                        (short)0, c1, false, false);
        }

        // ---- scale + mask + online softmax ----
        const int  keyg0 = kt * BK + ln;
        const int  keyg1 = keyg0 + 16;
        const bool msk0  = (long long)keyg0 >= vl;
        const bool msk1  = (long long)keyg1 >= vl;
#pragma unroll
        for (int r = 0; r < 8; ++r) {
            float s0 = msk0 ? -1.0e6f : c0[r] * scale;
            float s1 = msk1 ? -1.0e6f : c1[r] * scale;
            float mx = fmaxf(s0, s1);
#pragma unroll
            for (int off = 1; off < 16; off <<= 1)
                mx = fmaxf(mx, __shfl_xor(mx, off, 16));
            const float mnew = fmaxf(m_i[r], mx);
            const float p0 = __expf(s0 - mnew);
            const float p1 = __expf(s1 - mnew);
            float rs = p0 + p1;
#pragma unroll
            for (int off = 1; off < 16; off <<= 1)
                rs += __shfl_xor(rs, off, 16);
            const float alpha = __expf(m_i[r] - mnew);
            l_i[r] = l_i[r] * alpha + rs;
            m_i[r] = mnew;
#pragma unroll
            for (int t = 0; t < 8; ++t) o[t][r] *= alpha;
            sP[wave][r + 8*hf][ln]      = (_Float16)p0;
            sP[wave][r + 8*hf][16 + ln] = (_Float16)p1;
        }
        asm volatile("s_wait_dscnt 0" ::: "memory");  // wave-local LDS RAW

        // ---- P: C layout -> A layout (16x32 f16) ----
        v16h pa;
#pragma unroll
        for (int h = 0; h < 16; ++h) {
            const int v = h >> 1, p = h & 1;
            const int k = (v & 3)*2 + p + hf*8 + (v >> 2)*16;
            pa[h] = sP[wave][ln][k];
        }

        // ---- O += P * V (8 output d-chunks of 16) ----
#pragma unroll
        for (int t = 0; t < 8; ++t) {
            v16h vb;
#pragma unroll
            for (int h = 0; h < 16; ++h) {
                const int kd = (h & 7) + 8*hf + 16*(h >> 3);  // key within tile
                vb[h] = sVt[16*t + ln][kd];
            }
            o[t] = __builtin_amdgcn_wmma_f32_16x16x32_f16(false, pa, false, vb,
                                                          (short)0, o[t], false, false);
        }
        __syncthreads();  // protect sK/sVt before next tile load
    }

    // ---- epilogue: normalize by row sum, store fp32 ----
#pragma unroll
    for (int r = 0; r < 8; ++r) {
        const float inv = 1.0f / l_i[r];
        float* op = Out + ((long long)b * Lq + q0 + wave*16 + r + 8*hf) * DH + ln;
#pragma unroll
        for (int t = 0; t < 8; ++t)
            op[16*t] = o[t][r] * inv;
    }
}

extern "C" void kernel_launch(void* const* d_in, const int* in_sizes, int n_in,
                              void* d_out, int out_size, void* d_ws, size_t ws_size,
                              hipStream_t stream) {
    const float*     Q    = (const float*)d_in[0];
    const float*     K    = (const float*)d_in[1];
    const float*     V    = (const float*)d_in[2];
    const long long* vlen = (const long long*)d_in[3];  // jnp.int64
    float*           Out  = (float*)d_out;

    const int B  = in_sizes[3];
    const int Lq = in_sizes[0] / (B * DH);
    const int Lk = in_sizes[1] / (B * DH);

    dim3 grid(B * (Lq / BQ));
    fattn_f16wmma_kernel<<<grid, NT, 0, stream>>>(Q, K, V, vlen, Out, Lq, Lk);
}
